// GAT_Custom_36249523978301
// MI455X (gfx1250) — matrix operand order repaired and checked by reading, not compiled, not run on hardware
//
#include <hip/hip_runtime.h>
#include <hip/hip_bf16.h>

// ---------------------------------------------------------------------------
// Problem constants (from the reference)
// ---------------------------------------------------------------------------
#define GN   100000   // nodes
#define GE   800000   // edges (before self loops)
#define GIN  256
#define GH   4
#define GC   32
#define GF   128      // H*C
#define GOUT 64
#define GJK  320      // 128+128+64
#define NEG_SLOPE 0.2f

typedef __bf16 bf16_t;
typedef __attribute__((ext_vector_type(16))) __bf16 v16bf;
typedef __attribute__((ext_vector_type(8)))  __bf16 v8bf;
typedef __attribute__((ext_vector_type(8)))  float  v8f;

// ---------------------------------------------------------------------------
// fp32 -> bf16 conversion (row-major copy, used for activations)
// ---------------------------------------------------------------------------
__global__ void cvt_bf16_kernel(const float* __restrict__ in,
                                bf16_t* __restrict__ out, long long n) {
    long long i = (long long)blockIdx.x * blockDim.x + threadIdx.x;
    if (i < n) out[i] = (bf16_t)in[i];
}

// fp32 [M x K] -> bf16 transposed [K x M]  (weights; makes B-fragment loads
// contiguous 32B per lane for the WMMA layout)
__global__ void cvt_bf16_tr_kernel(const float* __restrict__ in,
                                   bf16_t* __restrict__ out, int M, int K) {
    long long o = (long long)blockIdx.x * blockDim.x + threadIdx.x;
    if (o >= (long long)M * K) return;
    int col = (int)(o / M);   // output row  (0..K-1)
    int k   = (int)(o % M);   // output col  (0..M-1)
    out[o] = (bf16_t)in[(size_t)k * K + col];
}

// ---------------------------------------------------------------------------
// WMMA GEMM:  C[Nrows x K] = A_bf16[Nrows x M] @ W^T_bf16[K x M] (+ bias[K])
// One wave computes a 16x64 output strip = 4 accumulator tiles; K-loop in
// steps of 32 via v_wmma_f32_16x16x32_bf16 (fp32 accumulate).
// Requires: Nrows%16==0, M%32==0, K%64==0 (all hold: M in {256,128,320},
// K in {128,64}).
//
// Fragment layouts (wave32, 16-bit WMMA):
//   A 16x32: lane lm = row; lane-half selects K sub-blocks
//            {half*16 .. half*16+7} and {half*16+8 .. half*16+15}  -> 2x b128
//   B 32x16: lane lm = col; lane-half selects K = half*16 .. half*16+15
//            -> contiguous 32B from transposed weights
//   D 16x16: VGPR r -> row r + half*8, col = lm
// ---------------------------------------------------------------------------
template <bool HAS_BIAS>
__global__ void wmma_gemm_kernel(const bf16_t* __restrict__ A,
                                 const bf16_t* __restrict__ Wt,
                                 const float* __restrict__ bias,
                                 float* __restrict__ C,
                                 int M, int K) {
    const int rowTile = blockIdx.x * 16;
    const int colBase = blockIdx.y * 64;       // 4 x 16-wide col tiles
    const int lane = threadIdx.x & 31;
    const int half = lane >> 4;
    const int lm   = lane & 15;

    v8f acc[4];
    #pragma unroll
    for (int t = 0; t < 4; ++t) acc[t] = (v8f){};

    const bf16_t* Arow = A + (size_t)(rowTile + lm) * M + half * 16;
    const bf16_t* Wrow = Wt + (size_t)(colBase + lm) * M + half * 16;

    for (int kb = 0; kb < M; kb += 32) {
        v8bf a_lo = *(const v8bf*)(Arow + kb);        // K = kb+half*16 .. +7
        v8bf a_hi = *(const v8bf*)(Arow + kb + 8);    // K = kb+half*16+8 .. +15
        v16bf a = __builtin_shufflevector(a_lo, a_hi,
            0, 1, 2, 3, 4, 5, 6, 7, 8, 9, 10, 11, 12, 13, 14, 15);
        #pragma unroll
        for (int t = 0; t < 4; ++t) {
            v16bf b = *(const v16bf*)(Wrow + (size_t)t * 16 * M + kb);
            acc[t] = __builtin_amdgcn_wmma_f32_16x16x32_bf16(
                /*neg_a=*/false, a, /*neg_b=*/false, b,
                /*c_mod=*/(short)0, acc[t], /*reuse_a=*/false, /*reuse_b=*/false);
        }
    }

    float bv[4];
    #pragma unroll
    for (int t = 0; t < 4; ++t)
        bv[t] = HAS_BIAS ? bias[colBase + t * 16 + lm] : 0.f;

    #pragma unroll
    for (int r = 0; r < 8; ++r) {
        const int row = rowTile + r + half * 8;
        float* Crow = C + (size_t)row * K + colBase + lm;
        #pragma unroll
        for (int t = 0; t < 4; ++t)
            Crow[t * 16] = acc[t][r] + bv[t];
    }
}

// ---------------------------------------------------------------------------
// Attention logits: al_src[n,h] = <xh[n,h,:], a_src[h,:]>, same for dst
// ---------------------------------------------------------------------------
__global__ void alpha_kernel(const float* __restrict__ xh,
                             const float* __restrict__ a_src,
                             const float* __restrict__ a_dst,
                             float* __restrict__ alsrc,
                             float* __restrict__ aldst,
                             int Nn, int Hh, int Cc) {
    int idx = blockIdx.x * blockDim.x + threadIdx.x;  // n*H + h
    if (idx >= Nn * Hh) return;
    int h = idx % Hh;
    const float* xp = xh + (size_t)idx * Cc;   // (n*H + h)*C
    float s = 0.f, d = 0.f;
    for (int c = 0; c < Cc; ++c) {
        float v = xp[c];
        s += v * a_src[h * Cc + c];
        d += v * a_dst[h * Cc + c];
    }
    alsrc[idx] = s;
    aldst[idx] = d;
}

// ---------------------------------------------------------------------------
// Per-layer init: emax = -inf, denom = 0, agg = 0
// ---------------------------------------------------------------------------
__global__ void init_kernel(float* __restrict__ emax, float* __restrict__ denom,
                            float* __restrict__ agg, long long NH, long long NF) {
    long long i = (long long)blockIdx.x * blockDim.x + threadIdx.x;
    if (i < NH) { emax[i] = __int_as_float(0xff800000); denom[i] = 0.f; }
    if (i < NF) agg[i] = 0.f;
}

// signed float atomic max via int/uint atomics (handles negatives; init -inf)
__device__ __forceinline__ void atomicMaxF(float* addr, float v) {
    if (v >= 0.f) atomicMax((int*)addr, __float_as_int(v));
    else          atomicMin((unsigned int*)addr, __float_as_uint(v));
}

__device__ __forceinline__ float leaky(float x) {
    return x > 0.f ? x : NEG_SLOPE * x;
}

// ---------------------------------------------------------------------------
// Segment max of leaky(al_src[s]+al_dst[d]) into emax[dst]; thread/(edge,head)
// Edges e in [0,E) come from edge_index; [E, E+N) are self loops.
// ---------------------------------------------------------------------------
__global__ void edge_max_kernel(const long long* __restrict__ ei,
                                const float* __restrict__ alsrc,
                                const float* __restrict__ aldst,
                                float* __restrict__ emax, int Hh) {
    long long t = (long long)blockIdx.x * blockDim.x + threadIdx.x;
    long long total = (long long)(GE + GN) * Hh;
    if (t >= total) return;
    int h = (int)(t % Hh);
    long long e = t / Hh;
    long long s, d;
    if (e < GE) { s = ei[e]; d = ei[GE + e]; } else { s = d = e - GE; }
    float el = leaky(alsrc[s * Hh + h] + aldst[d * Hh + h]);
    atomicMaxF(&emax[d * Hh + h], el);
}

// ---------------------------------------------------------------------------
// Edge accumulate: one wave per edge.
//   w[h] = exp(leaky(...) - emax[d,h]);  denom[d,h] += w[h]
//   agg[d, h*C + c] += xh[s, h*C + c] * w[h]   (lanes stride the F channels)
// ---------------------------------------------------------------------------
__global__ void edge_accum_kernel(const long long* __restrict__ ei,
                                  const float* __restrict__ alsrc,
                                  const float* __restrict__ aldst,
                                  const float* __restrict__ emax,
                                  const float* __restrict__ xh,
                                  float* __restrict__ denom,
                                  float* __restrict__ agg,
                                  int Hh, int Cc) {
    const int lane = threadIdx.x & 31;
    long long e = (long long)blockIdx.x * (blockDim.x >> 5) + (threadIdx.x >> 5);
    if (e >= (long long)(GE + GN)) return;
    long long s, d;
    if (e < GE) { s = ei[e]; d = ei[GE + e]; } else { s = d = e - GE; }

    float w[GH];
    for (int h = 0; h < Hh; ++h) {
        float el = leaky(alsrc[s * Hh + h] + aldst[d * Hh + h]);
        w[h] = __expf(el - emax[d * Hh + h]);
    }
    if (lane < Hh) atomicAdd(&denom[d * Hh + lane], w[lane]);

    const int F = Hh * Cc;
    for (int f = lane; f < F; f += 32) {
        int h = f / Cc;
        atomicAdd(&agg[d * (long long)F + f], xh[(size_t)s * F + f] * w[h]);
    }
}

// ---------------------------------------------------------------------------
// Finalize: out = agg/denom + bias + res, optional ELU; re-emit bf16 for the
// next layer's WMMA GEMM.
// ---------------------------------------------------------------------------
__global__ void finalize_kernel(const float* __restrict__ agg,
                                const float* __restrict__ denom,
                                const float* __restrict__ bias,
                                const float* __restrict__ res,
                                float* __restrict__ outf,
                                bf16_t* __restrict__ outbf,
                                int Hh, int Cc, int apply_elu) {
    const int F = Hh * Cc;
    long long idx = (long long)blockIdx.x * blockDim.x + threadIdx.x;
    if (idx >= (long long)GN * F) return;
    long long n = idx / F;
    int f = (int)(idx % F);
    int h = f / Cc;
    float v = agg[idx] / denom[n * Hh + h] + bias[f] + res[idx];
    if (apply_elu) v = v > 0.f ? v : (__expf(v) - 1.f);
    outf[idx] = v;
    if (outbf) outbf[idx] = (bf16_t)v;
}

// ---------------------------------------------------------------------------
// JumpingKnowledge concat [x0 | x1 | h2] -> bf16 [N, 320]
// ---------------------------------------------------------------------------
__global__ void jk_concat_kernel(const float* __restrict__ x0,
                                 const float* __restrict__ x1,
                                 const float* __restrict__ h2,
                                 bf16_t* __restrict__ jkb) {
    long long idx = (long long)blockIdx.x * blockDim.x + threadIdx.x;
    if (idx >= (long long)GN * GJK) return;
    long long n = idx / GJK;
    int c = (int)(idx % GJK);
    float v;
    if (c < GF)            v = x0[n * GF + c];
    else if (c < 2 * GF)   v = x1[n * GF + (c - GF)];
    else                   v = h2[n * GOUT + (c - 2 * GF)];
    jkb[idx] = (bf16_t)v;
}

// ---------------------------------------------------------------------------
// Host-side orchestration
// ---------------------------------------------------------------------------
static inline long long cdiv(long long a, long long b) { return (a + b - 1) / b; }

extern "C" void kernel_launch(void* const* d_in, const int* in_sizes, int n_in,
                              void* d_out, int out_size, void* d_ws, size_t ws_size,
                              hipStream_t stream) {
    const float*     x       = (const float*)d_in[0];
    const long long* ei      = (const long long*)d_in[1];
    const float*     W0      = (const float*)d_in[2];
    const float*     b0      = (const float*)d_in[3];
    const float*     as0     = (const float*)d_in[4];
    const float*     ad0     = (const float*)d_in[5];
    const float*     W1      = (const float*)d_in[6];
    const float*     b1      = (const float*)d_in[7];
    const float*     as1     = (const float*)d_in[8];
    const float*     ad1     = (const float*)d_in[9];
    const float*     W2      = (const float*)d_in[10];
    const float*     b2      = (const float*)d_in[11];
    const float*     as2     = (const float*)d_in[12];
    const float*     ad2     = (const float*)d_in[13];
    const float*     skip0_W = (const float*)d_in[14];
    const float*     skip0_b = (const float*)d_in[15];
    const float*     skip2_W = (const float*)d_in[16];
    const float*     skip2_b = (const float*)d_in[17];
    const float*     jk_W    = (const float*)d_in[18];
    const float*     jk_b    = (const float*)d_in[19];
    float* out = (float*)d_out;

    // -------- workspace carve-up --------
    char* ws = (char*)d_ws;
    size_t off = 0;
    auto alloc = [&](size_t bytes) -> void* {
        void* p = ws + off;
        off = (off + bytes + 255) & ~(size_t)255;
        return p;
    };
    bf16_t* featbf = (bf16_t*)alloc((size_t)GN * GIN * sizeof(bf16_t)); // reused per layer
    bf16_t* W0b    = (bf16_t*)alloc((size_t)GIN * GF * sizeof(bf16_t));   // transposed
    bf16_t* S0b    = (bf16_t*)alloc((size_t)GIN * GF * sizeof(bf16_t));   // transposed
    bf16_t* W1b    = (bf16_t*)alloc((size_t)GF * GF * sizeof(bf16_t));    // transposed
    bf16_t* W2b    = (bf16_t*)alloc((size_t)GF * GOUT * sizeof(bf16_t));  // transposed
    bf16_t* S2b    = (bf16_t*)alloc((size_t)GF * GOUT * sizeof(bf16_t));  // transposed
    bf16_t* JWb    = (bf16_t*)alloc((size_t)GJK * GOUT * sizeof(bf16_t)); // transposed
    float*  xh     = (float*)alloc((size_t)GN * GF * sizeof(float));
    float*  res    = (float*)alloc((size_t)GN * GF * sizeof(float));
    float*  agg    = (float*)alloc((size_t)GN * GF * sizeof(float));
    float*  x0     = (float*)alloc((size_t)GN * GF * sizeof(float));
    float*  x1     = (float*)alloc((size_t)GN * GF * sizeof(float));
    float*  h2     = (float*)alloc((size_t)GN * GOUT * sizeof(float));
    float*  alsrc  = (float*)alloc((size_t)GN * GH * sizeof(float));
    float*  aldst  = (float*)alloc((size_t)GN * GH * sizeof(float));
    float*  emax   = (float*)alloc((size_t)GN * GH * sizeof(float));
    float*  denom  = (float*)alloc((size_t)GN * GH * sizeof(float));
    bf16_t* jkbf   = (bf16_t*)alloc((size_t)GN * GJK * sizeof(bf16_t));
    (void)ws_size;

    const int TB = 256;
    auto cvt = [&](const float* src, bf16_t* dst, long long n) {
        cvt_bf16_kernel<<<(int)cdiv(n, TB), TB, 0, stream>>>(src, dst, n);
    };
    auto cvt_tr = [&](const float* src, bf16_t* dst, int M, int K) {
        cvt_bf16_tr_kernel<<<(int)cdiv((long long)M * K, TB), TB, 0, stream>>>(
            src, dst, M, K);
    };

    // -------- weight / input conversion (fp32 -> bf16; weights transposed) --
    cvt(x, featbf, (long long)GN * GIN);
    cvt_tr(W0,      W0b, GIN, GF);
    cvt_tr(skip0_W, S0b, GIN, GF);
    cvt_tr(W1,      W1b, GF,  GF);
    cvt_tr(W2,      W2b, GF,  GOUT);
    cvt_tr(skip2_W, S2b, GF,  GOUT);
    cvt_tr(jk_W,    JWb, GJK, GOUT);

    const long long nEdgeAll = (long long)(GE + GN);
    const int wavesPerBlk = TB / 32;

    auto gemm = [&](const bf16_t* A, const bf16_t* Wt, const float* bias,
                    float* C, int M, int K) {
        dim3 g(GN / 16, K / 64);
        if (bias)
            wmma_gemm_kernel<true><<<g, 32, 0, stream>>>(A, Wt, bias, C, M, K);
        else
            wmma_gemm_kernel<false><<<g, 32, 0, stream>>>(A, Wt, nullptr, C, M, K);
    };

    auto gat_layer = [&](const bf16_t* feat, int Min, const bf16_t* Wb,
                         const float* b, const float* asrc_w, const float* adst_w,
                         int Hh, int Cc, const float* resid, float* outF,
                         bf16_t* outB, int elu) {
        const int F = Hh * Cc;
        // feature transform on WMMA
        gemm(feat, Wb, nullptr, xh, Min, F);
        // attention logits
        alpha_kernel<<<(int)cdiv((long long)GN * Hh, TB), TB, 0, stream>>>(
            xh, asrc_w, adst_w, alsrc, aldst, GN, Hh, Cc);
        // softmax state init
        init_kernel<<<(int)cdiv((long long)GN * F, TB), TB, 0, stream>>>(
            emax, denom, agg, (long long)GN * Hh, (long long)GN * F);
        // segment max
        edge_max_kernel<<<(int)cdiv(nEdgeAll * Hh, TB), TB, 0, stream>>>(
            ei, alsrc, aldst, emax, Hh);
        // weighted aggregation (wave per edge)
        edge_accum_kernel<<<(int)cdiv(nEdgeAll, wavesPerBlk), TB, 0, stream>>>(
            ei, alsrc, aldst, emax, xh, denom, agg, Hh, Cc);
        // normalize + bias + residual + activation, emit next-layer bf16
        finalize_kernel<<<(int)cdiv((long long)GN * F, TB), TB, 0, stream>>>(
            agg, denom, b, resid, outF, outB, Hh, Cc, elu);
    };

    // -------- layer 0: 256 -> 4x32 concat, skip = Linear(256->128) --------
    gemm(featbf, S0b, skip0_b, res, GIN, GF);
    gat_layer(featbf, GIN, W0b, b0, as0, ad0, GH, GC, res, x0, featbf, 1);

    // -------- layer 1: 128 -> 4x32 concat, skip = identity (x0) --------
    gat_layer(featbf, GF, W1b, b1, as1, ad1, GH, GC, x0, x1, featbf, 1);

    // -------- layer 2: 128 -> 1x64 mean, skip = Linear(128->64), no ELU ----
    gemm(featbf, S2b, skip2_b, res, GF, GOUT);
    gat_layer(featbf, GF, W2b, b2, as2, ad2, 1, GOUT, res, h2, nullptr, 0);

    // -------- JumpingKnowledge cat + final linear --------
    jk_concat_kernel<<<(int)cdiv((long long)GN * GJK, TB), TB, 0, stream>>>(
        x0, x1, h2, jkbf);
    gemm(jkbf, JWb, jk_b, out, GJK, GOUT);
}
